// BasicsTransformerLM_49813030699406
// MI455X (gfx1250) — matrix operand order repaired and checked by reading, not compiled, not run on hardware
//
#include <hip/hip_runtime.h>
#include <hip/hip_bf16.h>

// ---------------------------------------------------------------------------
// MI455X (gfx1250) transformer forward, all matmuls on v_wmma_f32_16x16x32_bf16.
//  - Weights pre-converted fp32->bf16 once per launch; GEMM hot loop is pure
//    b128 loads + 8 WMMAs per K32 step (last iteration peeled so the weight
//    prefetch is branch-free).
//  - Q/K/V bf16 post-RoPE; V additionally transposed to [b,h,d,s] so flash
//    attention's P*V fragments are contiguous 32B loads.
//  - Flash attention: one wave per (b,h,16-row q tile), online softmax.
// ---------------------------------------------------------------------------

#define D_MODEL 1024
#define N_HEADS 16
#define DK      64
#define DFF_    4096
#define SEQ     1024
#define BATCH   2
#define VOCAB_  32000
#define N_LAYERS 4
#define EPS_    1e-6f

typedef __attribute__((ext_vector_type(16))) __bf16 v16bf;
typedef __attribute__((ext_vector_type(8)))  float  v8f;

static __device__ __forceinline__ unsigned short f2bf_bits(float f) {
  union { float f; unsigned u; } c; c.f = f;
  unsigned r = c.u + 0x7fffu + ((c.u >> 16) & 1u);   // round-to-nearest-even
  return (unsigned short)(r >> 16);
}
static __device__ __forceinline__ __bf16 bits2bf(unsigned short u) {
  return __builtin_bit_cast(__bf16, u);
}

// ---------------------------------------------------------------------------
// h[bs, d] = emb[x[bs], d]
// ---------------------------------------------------------------------------
__global__ void embed_kernel(const int* __restrict__ x,
                             const float* __restrict__ emb,
                             float* __restrict__ h) {
  int idx = blockIdx.x * blockDim.x + threadIdx.x;      // over M*D
  int bs = idx >> 10;                                   // D_MODEL == 1024
  int d  = idx & (D_MODEL - 1);
  h[idx] = emb[(size_t)x[bs] * D_MODEL + d];
}

// ---------------------------------------------------------------------------
// bulk f32 -> bf16 with scale (weights: scale=1; Q: scale=1/8)
// ---------------------------------------------------------------------------
__global__ void cvt_bf16_kernel(const float* __restrict__ in,
                                unsigned short* __restrict__ out,
                                int n, float scale) {
  int i = blockIdx.x * blockDim.x + threadIdx.x;
  if (i < n) out[i] = f2bf_bits(in[i] * scale);
}

// ---------------------------------------------------------------------------
// vt[b,h,d,s] = bf16(v[b,s,h*DK+d])  — key-major V for contiguous B fragments
// idx bits: s:10 | d:6 | h:4 | b:1
// ---------------------------------------------------------------------------
__global__ void v_transpose_kernel(const float* __restrict__ v,
                                   unsigned short* __restrict__ vt) {
  const int idx = blockIdx.x * blockDim.x + threadIdx.x;
  const int s = idx & (SEQ - 1);
  const int d = (idx >> 10) & (DK - 1);
  const int h = (idx >> 16) & (N_HEADS - 1);
  const int b = idx >> 20;
  vt[idx] = f2bf_bits(v[((size_t)(b * SEQ + s)) * D_MODEL + h * DK + d]);
}

// ---------------------------------------------------------------------------
// out_bf16[row, :] = w[:] * x[row, :] * rsqrt(mean(x^2) + eps)
// ---------------------------------------------------------------------------
__global__ void rmsnorm_bf16_kernel(const float* __restrict__ x,
                                    const float* __restrict__ w,
                                    unsigned short* __restrict__ out) {
  __shared__ float red[256];
  const int row = blockIdx.x;
  const float* xr = x + (size_t)row * D_MODEL;
  float s = 0.f;
  for (int d = threadIdx.x; d < D_MODEL; d += 256) { float v = xr[d]; s += v * v; }
  red[threadIdx.x] = s;
  __syncthreads();
  for (int off = 128; off > 0; off >>= 1) {
    if (threadIdx.x < off) red[threadIdx.x] += red[threadIdx.x + off];
    __syncthreads();
  }
  const float r = rsqrtf(red[0] * (1.0f / D_MODEL) + EPS_);
  for (int d = threadIdx.x; d < D_MODEL; d += 256)
    out[(size_t)row * D_MODEL + d] = f2bf_bits(w[d] * xr[d] * r);
}

// ---------------------------------------------------------------------------
// GEMM (NT): C[M,N] (+)= A_bf16[M,K] * W_bf16[N,K]^T   (fp32 accumulate)
// Block = 8 waves; wave tile = 32(M) x 64(N); K step 32 -> 8 WMMAs / step.
// Last K iteration peeled so the main loop's weight prefetch is branch-free.
// ---------------------------------------------------------------------------
__global__ void gemm_bf16_nt_kernel(const unsigned short* __restrict__ A,
                                    const unsigned short* __restrict__ W,
                                    float* __restrict__ C,
                                    int N, int K, int accumulate) {
  const int lane = threadIdx.x & 31;
  const int wave = threadIdx.x >> 5;
  const int hi   = lane >> 4;
  const int nlo  = lane & 15;
  const int rowBase = blockIdx.x * 256 + wave * 32;
  const int colBase = blockIdx.y * 64;
  const int akb  = 16 * hi;

  v8f acc[2][4] = {};

  auto kstep = [&](int k0, bool prefetch) {
    v16bf a[2];
#pragma unroll
    for (int mt = 0; mt < 2; ++mt) {
      const unsigned short* ap = A + (size_t)(rowBase + mt * 16 + nlo) * K + k0 + akb;
#pragma unroll
      for (int i = 0; i < 8; ++i) { a[mt][i] = bits2bf(ap[i]); a[mt][i + 8] = bits2bf(ap[8 + i]); }
    }
    v16bf b[4];
#pragma unroll
    for (int nt = 0; nt < 4; ++nt) {
      const unsigned short* wp = W + (size_t)(colBase + nt * 16 + nlo) * K + k0 + 16 * hi;
      if (prefetch) __builtin_prefetch(wp + 32, 0, 3);   // next K slice, branch-free
#pragma unroll
      for (int i = 0; i < 16; ++i) b[nt][i] = bits2bf(wp[i]);
    }
#pragma unroll
    for (int mt = 0; mt < 2; ++mt)
#pragma unroll
      for (int nt = 0; nt < 4; ++nt)
        acc[mt][nt] = __builtin_amdgcn_wmma_f32_16x16x32_bf16(
            false, a[mt], false, b[nt], (short)0, acc[mt][nt], false, false);
  };

  for (int k0 = 0; k0 < K - 32; k0 += 32) kstep(k0, true);
  kstep(K - 32, false);

#pragma unroll
  for (int mt = 0; mt < 2; ++mt)
#pragma unroll
    for (int nt = 0; nt < 4; ++nt)
#pragma unroll
      for (int r = 0; r < 8; ++r) {
        const int row = rowBase + mt * 16 + r + 8 * hi;
        const int col = colBase + nt * 16 + nlo;
        const size_t o = (size_t)row * N + col;
        const float v = acc[mt][nt][r];
        C[o] = accumulate ? (C[o] + v) : v;
      }
}

// ---------------------------------------------------------------------------
// RoPE in-place on q and k (fp32). One thread per (b, s, h, pair j).
// idx bits: j:5 | h:4 | s:10 | b:1
// ---------------------------------------------------------------------------
__global__ void rope_kernel(float* __restrict__ q, float* __restrict__ k) {
  const int idx = blockIdx.x * blockDim.x + threadIdx.x;
  const int j = idx & 31;
  const int h = (idx >> 5) & (N_HEADS - 1);
  const int s = (idx >> 9) & (SEQ - 1);
  const int b = idx >> 19;
  const float inv = powf(10000.f, -(float)j * (1.f / 32.f));
  float sn, cs;
  sincosf((float)s * inv, &sn, &cs);
  const size_t o = ((size_t)(b * SEQ + s)) * D_MODEL + h * DK + 2 * j;
  const float q1 = q[o], q2 = q[o + 1];
  q[o]     = q1 * cs - q2 * sn;
  q[o + 1] = q1 * sn + q2 * cs;
  const float k1 = k[o], k2 = k[o + 1];
  k[o]     = k1 * cs - k2 * sn;
  k[o + 1] = k1 * sn + k2 * cs;
}

// ---------------------------------------------------------------------------
// Flash attention (bf16 in / bf16 out, 1/8 scale pre-folded into qbf):
// one wave per (b, h, 16-row q tile). Scores: 2 WMMAs per 16x16 tile (dk=64).
// Causal online softmax on the C-fragment layout; P bounced through LDS to
// A layout; O += P*V with key-major Vt giving contiguous 32B B fragments.
// ---------------------------------------------------------------------------
__global__ void flash_attn_kernel(const unsigned short* __restrict__ q,
                                  const unsigned short* __restrict__ k,
                                  const unsigned short* __restrict__ vt,
                                  unsigned short* __restrict__ o) {
  __shared__ unsigned short pbuf[16 * 32];
  const int lane = threadIdx.x & 31;
  const int hi   = lane >> 4;
  const int nlo  = lane & 15;
  const int bid  = blockIdx.x;                 // b*H*(S/16) blocks
  const int qt   = bid & (SEQ / 16 - 1);
  const int h    = (bid >> 6) & (N_HEADS - 1);
  const int b    = bid >> 10;
  const int qbase = qt * 16;
  const size_t base  = ((size_t)b * SEQ) * D_MODEL + h * DK;          // q/k/o: + s*D + d
  const size_t vbase = ((size_t)(b * N_HEADS + h)) * DK * SEQ;        // vt:   + d*SEQ + s

  v16bf qf[2];
  {
    const unsigned short* qp = q + base + (size_t)(qbase + nlo) * D_MODEL;
#pragma unroll
    for (int f = 0; f < 2; ++f) {
      const int kb = 32 * f + 16 * hi;
#pragma unroll
      for (int i = 0; i < 8; ++i) {
        qf[f][i]     = bits2bf(qp[kb + i]);
        qf[f][i + 8] = bits2bf(qp[kb + 8 + i]);
      }
    }
  }

  v8f oa[4] = {};
  float mrun[8], lrun[8];
#pragma unroll
  for (int r = 0; r < 8; ++r) { mrun[r] = -1e30f; lrun[r] = 0.f; }

  const int ntiles = (qbase + 16 + 31) / 32;
  for (int t = 0; t < ntiles; ++t) {
    const int k0 = t * 32;

    v8f sc[2] = {};
#pragma unroll
    for (int st = 0; st < 2; ++st) {
      const unsigned short* kp = k + base + (size_t)(k0 + st * 16 + nlo) * D_MODEL;
      v16bf kf0, kf1;
#pragma unroll
      for (int i = 0; i < 16; ++i) {
        kf0[i] = bits2bf(kp[16 * hi + i]);
        kf1[i] = bits2bf(kp[32 + 16 * hi + i]);
      }
      sc[st] = __builtin_amdgcn_wmma_f32_16x16x32_bf16(
          false, qf[0], false, kf0, (short)0, sc[st], false, false);
      sc[st] = __builtin_amdgcn_wmma_f32_16x16x32_bf16(
          false, qf[1], false, kf1, (short)0, sc[st], false, false);
    }

    float p0a[8], p1a[8], fac[8];
#pragma unroll
    for (int r = 0; r < 8; ++r) {
      const int row = qbase + r + 8 * hi;
      float x0 = (k0 + nlo)      > row ? -1e30f : sc[0][r];
      float x1 = (k0 + 16 + nlo) > row ? -1e30f : sc[1][r];
      float mx = fmaxf(x0, x1);
#pragma unroll
      for (int off = 1; off < 16; off <<= 1)
        mx = fmaxf(mx, __shfl_xor(mx, off, 32));
      const float mnew = fmaxf(mrun[r], mx);
      const float f  = __expf(mrun[r] - mnew);
      const float e0 = __expf(x0 - mnew);
      const float e1 = __expf(x1 - mnew);
      float rs = e0 + e1;
#pragma unroll
      for (int off = 1; off < 16; off <<= 1)
        rs += __shfl_xor(rs, off, 32);
      lrun[r] = lrun[r] * f + rs;
      mrun[r] = mnew;
      fac[r] = f; p0a[r] = e0; p1a[r] = e1;
    }
#pragma unroll
    for (int r = 0; r < 8; ++r) {
      oa[0][r] *= fac[r]; oa[1][r] *= fac[r];
      oa[2][r] *= fac[r]; oa[3][r] *= fac[r];
    }

    __syncthreads();
#pragma unroll
    for (int r = 0; r < 8; ++r) {
      const int row = r + 8 * hi;
      pbuf[row * 32 + nlo]      = f2bf_bits(p0a[r]);
      pbuf[row * 32 + 16 + nlo] = f2bf_bits(p1a[r]);
    }
    __syncthreads();
    v16bf pf;
    {
      const unsigned short* pp = pbuf + nlo * 32 + 16 * hi;
#pragma unroll
      for (int i = 0; i < 8; ++i) { pf[i] = bits2bf(pp[i]); pf[i + 8] = bits2bf(pp[8 + i]); }
    }

#pragma unroll
    for (int nt = 0; nt < 4; ++nt) {
      v16bf vf;
      const unsigned short* vp = vt + vbase + (size_t)(nt * 16 + nlo) * SEQ + k0 + 16 * hi;
#pragma unroll
      for (int i = 0; i < 16; ++i) vf[i] = bits2bf(vp[i]);     // contiguous 32B
      oa[nt] = __builtin_amdgcn_wmma_f32_16x16x32_bf16(
          false, pf, false, vf, (short)0, oa[nt], false, false);
    }
  }

  float inv_l[8];
#pragma unroll
  for (int r = 0; r < 8; ++r) inv_l[r] = 1.f / lrun[r];
#pragma unroll
  for (int nt = 0; nt < 4; ++nt)
#pragma unroll
    for (int r = 0; r < 8; ++r) {
      const int srow = qbase + r + 8 * hi;
      o[base + (size_t)srow * D_MODEL + nt * 16 + nlo] =
          f2bf_bits(oa[nt][r] * inv_l[r]);
    }
}

// ---------------------------------------------------------------------------
// gu_bf16[i] = silu(gate[i]) * up[i]
// ---------------------------------------------------------------------------
__global__ void silu_mul_kernel(const float* __restrict__ gate,
                                const float* __restrict__ up,
                                unsigned short* __restrict__ out, int n) {
  int i = blockIdx.x * blockDim.x + threadIdx.x;
  if (i < n) {
    const float g = gate[i];
    const float s = g / (1.f + __expf(-g));
    out[i] = f2bf_bits(s * up[i]);
  }
}

// ---------------------------------------------------------------------------
// host orchestration
// input order (dict order): x, emb, [wq wk wv wo w1 w2 w3 ln1 ln2]x4, ln_f, lm_head
// ---------------------------------------------------------------------------
extern "C" void kernel_launch(void* const* d_in, const int* in_sizes, int n_in,
                              void* d_out, int out_size, void* d_ws, size_t ws_size,
                              hipStream_t stream) {
  (void)in_sizes; (void)n_in; (void)out_size; (void)ws_size;
  const int M = BATCH * SEQ;

  const int*   x    = (const int*)d_in[0];
  const float* emb  = (const float*)d_in[1];
  const float* ln_f = (const float*)d_in[2 + 9 * N_LAYERS + 0];
  const float* w_lm = (const float*)d_in[2 + 9 * N_LAYERS + 1];

  // workspace layout (~160 MB)
  char* ws = (char*)d_ws;
  size_t off = 0;
  auto alloc = [&](size_t bytes) -> void* {
    void* p = ws + off; off += (bytes + 255) & ~(size_t)255; return p;
  };
  float*          hbuf = (float*)alloc((size_t)M * D_MODEL * 4);
  unsigned short* abf  = (unsigned short*)alloc((size_t)M * D_MODEL * 2); // hn / O (bf16)
  float*          qb   = (float*)alloc((size_t)M * D_MODEL * 4);
  float*          kb   = (float*)alloc((size_t)M * D_MODEL * 4);
  float*          vb   = (float*)alloc((size_t)M * D_MODEL * 4);
  unsigned short* qbf  = (unsigned short*)alloc((size_t)M * D_MODEL * 2);
  unsigned short* kbf  = (unsigned short*)alloc((size_t)M * D_MODEL * 2);
  unsigned short* vtbf = (unsigned short*)alloc((size_t)M * D_MODEL * 2); // V^T [b,h,d,s]
  float*          gate = (float*)alloc((size_t)M * DFF_ * 4);   // } contiguous 64MB,
  float*          up   = (float*)alloc((size_t)M * DFF_ * 4);   // } reused for lm_head bf16
  unsigned short* gubf = (unsigned short*)alloc((size_t)M * DFF_ * 2);
  unsigned short* wbf  = (unsigned short*)alloc((size_t)16 * 1024 * 1024 * 2); // per-layer weights
  unsigned short* lmbf = (unsigned short*)gate;                 // 64MB alias (gate+up dead by then)

  const int szDD = D_MODEL * D_MODEL;   // 1M
  const int szDF = D_MODEL * DFF_;      // 4M
  unsigned short* wq_bf = wbf;
  unsigned short* wk_bf = wbf + 1 * szDD;
  unsigned short* wv_bf = wbf + 2 * szDD;
  unsigned short* wo_bf = wbf + 3 * szDD;
  unsigned short* w1_bf = wbf + 4 * szDD;
  unsigned short* w3_bf = w1_bf + szDF;
  unsigned short* w2_bf = w3_bf + szDF;

  embed_kernel<<<(M * D_MODEL) / 256, 256, 0, stream>>>(x, emb, hbuf);

  const dim3 gD (M / 256, D_MODEL / 64);
  const dim3 gF (M / 256, DFF_    / 64);
  const dim3 gV (M / 256, VOCAB_  / 64);

  for (int l = 0; l < N_LAYERS; ++l) {
    const float* wq  = (const float*)d_in[2 + 9 * l + 0];
    const float* wk  = (const float*)d_in[2 + 9 * l + 1];
    const float* wv  = (const float*)d_in[2 + 9 * l + 2];
    const float* wo  = (const float*)d_in[2 + 9 * l + 3];
    const float* w1  = (const float*)d_in[2 + 9 * l + 4];
    const float* w2  = (const float*)d_in[2 + 9 * l + 5];
    const float* w3  = (const float*)d_in[2 + 9 * l + 6];
    const float* ln1 = (const float*)d_in[2 + 9 * l + 7];
    const float* ln2 = (const float*)d_in[2 + 9 * l + 8];

    // cold pass: weights fp32 -> bf16
    cvt_bf16_kernel<<<szDD / 256, 256, 0, stream>>>(wq, wq_bf, szDD, 1.f);
    cvt_bf16_kernel<<<szDD / 256, 256, 0, stream>>>(wk, wk_bf, szDD, 1.f);
    cvt_bf16_kernel<<<szDD / 256, 256, 0, stream>>>(wv, wv_bf, szDD, 1.f);
    cvt_bf16_kernel<<<szDD / 256, 256, 0, stream>>>(wo, wo_bf, szDD, 1.f);
    cvt_bf16_kernel<<<szDF / 256, 256, 0, stream>>>(w1, w1_bf, szDF, 1.f);
    cvt_bf16_kernel<<<szDF / 256, 256, 0, stream>>>(w3, w3_bf, szDF, 1.f);
    cvt_bf16_kernel<<<szDF / 256, 256, 0, stream>>>(w2, w2_bf, szDF, 1.f);

    // attention block
    rmsnorm_bf16_kernel<<<M, 256, 0, stream>>>(hbuf, ln1, abf);
    gemm_bf16_nt_kernel<<<gD, 256, 0, stream>>>(abf, wq_bf, qb, D_MODEL, D_MODEL, 0);
    gemm_bf16_nt_kernel<<<gD, 256, 0, stream>>>(abf, wk_bf, kb, D_MODEL, D_MODEL, 0);
    gemm_bf16_nt_kernel<<<gD, 256, 0, stream>>>(abf, wv_bf, vb, D_MODEL, D_MODEL, 0);
    rope_kernel<<<(M * N_HEADS * 32) / 256, 256, 0, stream>>>(qb, kb);
    cvt_bf16_kernel<<<(M * D_MODEL) / 256, 256, 0, stream>>>(qb, qbf, M * D_MODEL, 0.125f);
    cvt_bf16_kernel<<<(M * D_MODEL) / 256, 256, 0, stream>>>(kb, kbf, M * D_MODEL, 1.f);
    v_transpose_kernel<<<(M * D_MODEL) / 256, 256, 0, stream>>>(vb, vtbf);
    flash_attn_kernel<<<BATCH * N_HEADS * (SEQ / 16), 32, 0, stream>>>(qbf, kbf, vtbf, abf);
    gemm_bf16_nt_kernel<<<gD, 256, 0, stream>>>(abf, wo_bf, hbuf, D_MODEL, D_MODEL, 1); // residual

    // SwiGLU FFN
    rmsnorm_bf16_kernel<<<M, 256, 0, stream>>>(hbuf, ln2, abf);
    gemm_bf16_nt_kernel<<<gF, 256, 0, stream>>>(abf, w1_bf, gate, DFF_, D_MODEL, 0);
    gemm_bf16_nt_kernel<<<gF, 256, 0, stream>>>(abf, w3_bf, up,   DFF_, D_MODEL, 0);
    silu_mul_kernel<<<(M * DFF_) / 256, 256, 0, stream>>>(gate, up, gubf, M * DFF_);
    gemm_bf16_nt_kernel<<<gD, 256, 0, stream>>>(gubf, w2_bf, hbuf, D_MODEL, DFF_, 1);  // residual
  }

  // final norm + lm_head (convert lm_head into dead gate/up region)
  rmsnorm_bf16_kernel<<<M, 256, 0, stream>>>(hbuf, ln_f, abf);
  cvt_bf16_kernel<<<(VOCAB_ * D_MODEL) / 256, 256, 0, stream>>>(w_lm, lmbf, VOCAB_ * D_MODEL, 1.f);
  gemm_bf16_nt_kernel<<<gV, 256, 0, stream>>>(abf, lmbf, (float*)d_out, VOCAB_, D_MODEL, 0);
}
